// MultiHeadAttention_78151224918580
// MI455X (gfx1250) — matrix operand order repaired and checked by reading, not compiled, not run on hardware
//
#include <hip/hip_runtime.h>
#include <hip/hip_bf16.h>
#include <stdint.h>

// ---------------------------------------------------------------------------
// MHA forward for B=2, S=2048, F=1024, H=16, D=64 on gfx1250 (MI455X).
// All matmuls via v_wmma_f32_16x16x32_bf16 (fp32 accumulate).
// QKV GEMM stages its weight tile with the Tensor Data Mover (TDM).
// Workspace layout (32 MB total):
//   [ 0MB.. 8MB)  Q   bf16 [B,H,S,D]   (pre-scaled by 1/sqrt(D))
//   [ 8MB..16MB)  K   bf16 [B,H,S,D]
//   [16MB..24MB)  V^T bf16 [B,H,D,S]
//   [24MB..32MB)  Z   bf16 [B,S,F]     (merged attention output)
// ---------------------------------------------------------------------------

#define DEVINL __device__ __forceinline__

typedef __attribute__((ext_vector_type(16))) __bf16 v16bf;
typedef __attribute__((ext_vector_type(8)))  __bf16 v8bf;
typedef __attribute__((ext_vector_type(8)))  float  v8f;
typedef __attribute__((ext_vector_type(4)))  float  v4f;
typedef __attribute__((ext_vector_type(4)))  unsigned int u32x4;
typedef __attribute__((ext_vector_type(8)))  int    i32x8;
typedef __attribute__((ext_vector_type(4)))  int    i32x4;

static constexpr int Bc = 2, Sc = 2048, Fc = 1024, Hc = 16, Dc = 64;
static constexpr int N3F = 3 * Fc;          // 3072

DEVINL v8f wmma_bf16(v16bf a, v16bf b, v8f c) {
  return __builtin_amdgcn_wmma_f32_16x16x32_bf16(false, a, false, b,
                                                 (short)0, c, false, false);
}

// Load 8 consecutive fp32 and convert to 8 bf16 (v_cvt_pk_bf16_f32 pairs).
DEVINL v8bf load_cvt8(const float* __restrict__ p) {
  v4f a = *(const v4f*)p;
  v4f b = *(const v4f*)(p + 4);
  v8bf r;
#pragma unroll
  for (int i = 0; i < 4; ++i) { r[i] = (__bf16)a[i]; r[i + 4] = (__bf16)b[i]; }
  return r;
}

// A-fragment (16x32 bf16) from a row-major fp32 matrix.
// Contiguous K order used here (B matches): lane L -> row M = L&15; g = L>>4; elem i -> K = 16g+i.
DEVINL v16bf a_frag_f32(const float* __restrict__ rowp, int k0, int g) {
  const float* p = rowp + k0 + 16 * g;
  v8bf lo = load_cvt8(p);
  v8bf hi = load_cvt8(p + 8);
  v16bf r;
#pragma unroll
  for (int i = 0; i < 8; ++i) { r[i] = lo[i]; r[8 + i] = hi[i]; }
  return r;
}

// A-fragment (16x32) from a row-major bf16 matrix (contiguous 16B chunks).
DEVINL v16bf a_frag_bf16(const __bf16* __restrict__ rowp, int k0, int g) {
  v8bf lo = *(const v8bf*)(rowp + k0 + 16 * g);
  v8bf hi = *(const v8bf*)(rowp + k0 + 16 * g + 8);
  v16bf r;
#pragma unroll
  for (int i = 0; i < 8; ++i) { r[i] = lo[i]; r[8 + i] = hi[i]; }
  return r;
}

// B-fragment (32x16) from an fp32 [32][64] LDS tile (TDM-staged, row-major).
// Contiguous K order used here (matches A): lane N = L&15 (+ sub offset), elems e -> K = 16g + e.
DEVINL v16bf b_frag_lds_f32(const float* lds_tile, int n, int g) {
  const float* p = lds_tile + (size_t)(16 * g) * 64 + n;
  v16bf r;
#pragma unroll
  for (int e = 0; e < 16; ++e) r[e] = (__bf16)p[e * 64];
  return r;
}

// ---------------------------------------------------------------------------
// Kernel 1: QKV projection.  X[4096,1024] (fp32) @ W[1024,3072] (fp32) + bias
// -> Q (scaled, bf16 [B,H,S,D]), K (bf16 [B,H,S,D]), V^T (bf16 [B,H,D,S]).
// The 32x64 fp32 W tile is DMA'd global->LDS by the Tensor Data Mover each
// K-step (wave 0 issues, s_wait_tensorcnt 0, workgroup barrier), then all 8
// waves build bf16 B-fragments from the LDS tile.
// ---------------------------------------------------------------------------
__global__ __launch_bounds__(256)
void qkv_gemm_kernel(const float* __restrict__ X, const float* __restrict__ W,
                     const float* __restrict__ bias,
                     __bf16* __restrict__ q, __bf16* __restrict__ kbuf,
                     __bf16* __restrict__ vT) {
  __shared__ __align__(16) float wtile[32 * 64];   // 8KB TDM destination

  const int lane = threadIdx.x & 31;
  const int wv   = threadIdx.x >> 5;
  const int g    = lane >> 4;
  const int ln   = lane & 15;
  const int m0   = (blockIdx.y * 8 + wv) * 16;
  const int n0   = blockIdx.x * 64;

  const float* arow = X + (size_t)(m0 + ln) * Fc;
  // LDS byte offset = low 32 bits of the generic (flat-aperture) address.
  const unsigned lds_base = (unsigned)(uintptr_t)&wtile[0];

  const v8f zero = {0.f, 0.f, 0.f, 0.f, 0.f, 0.f, 0.f, 0.f};
  v8f acc[4] = {zero, zero, zero, zero};

  for (int k0 = 0; k0 < Fc; k0 += 32) {
    // ---- TDM: stage W[k0..k0+32)[n0..n0+64) (fp32) into LDS, wave 0 only
    if (wv == 0) {
      const unsigned long long gaddr =
          (unsigned long long)(uintptr_t)(W + (size_t)k0 * N3F + n0);
      // D# group 0: count=1, lds_addr, 57-bit global tile address, type=2.
      u32x4 g0 = {1u, lds_base, (unsigned)(gaddr & 0xffffffffu),
                  (unsigned)((gaddr >> 32) & 0x1ffffffu) | 0x80000000u};
      // D# group 1: data_size=4B, tensor 3072x1024, tile 64x32, stride 3072.
      i32x8 g1 = {(int)0x00020000u,                         // data_size=2 (4B)
                  (int)(((unsigned)N3F & 0xffffu) << 16),   // tensor_dim0 lo16
                  (int)((((unsigned)N3F >> 16) & 0xffffu) |
                        (((unsigned)Fc & 0xffffu) << 16)),  // dim0 hi | dim1 lo
                  (int)((((unsigned)Fc >> 16) & 0xffffu) |
                        (64u << 16)),                       // dim1 hi | tile_dim0
                  32,                                       // tile_dim1 (| tile_dim2=0)
                  N3F,                                      // dim0_stride lo32
                  0, 0};                                    // stride hi, dim1_stride
      i32x4 gz = {0, 0, 0, 0};
#if defined(__clang_major__) && __clang_major__ >= 23
      i32x8 gz8 = {0, 0, 0, 0, 0, 0, 0, 0};
      __builtin_amdgcn_tensor_load_to_lds(g0, g1, gz, gz, gz8, 0);
#else
      __builtin_amdgcn_tensor_load_to_lds(g0, g1, gz, gz, 0);
#endif
      __builtin_amdgcn_s_wait_tensorcnt((short)0);
    }
    __syncthreads();

    if (k0 + 32 < Fc) __builtin_prefetch(arow + k0 + 32, 0, 3);

    v16bf af = a_frag_f32(arow, k0, g);
#pragma unroll
    for (int sub = 0; sub < 4; ++sub) {
      v16bf bf = b_frag_lds_f32(wtile, sub * 16 + ln, g);
      acc[sub] = wmma_bf16(af, bf, acc[sub]);
    }
    __syncthreads();
  }

  // --- epilogue: bias + scatter to Q / K / V^T in bf16
  const int sec = n0 >> 10;             // 0 = Q, 1 = K, 2 = V
  const int h   = (n0 & 1023) >> 6;     // head index (tile never crosses heads)
#pragma unroll
  for (int sub = 0; sub < 4; ++sub) {
    const int   dcol = sub * 16 + ln;   // 0..63 within the head
    const float bv   = bias[n0 + dcol];
#pragma unroll
    for (int r = 0; r < 8; ++r) {
      const int   m = m0 + r + 8 * g;   // C layout: M = r + 8g, N = ln
      const int   bb = m >> 11, ss = m & (Sc - 1);
      const float val = acc[sub][r] + bv;
      if (sec == 0) {
        q[((size_t)(bb * Hc + h) * Sc + ss) * Dc + dcol] = (__bf16)(val * 0.125f);
      } else if (sec == 1) {
        kbuf[((size_t)(bb * Hc + h) * Sc + ss) * Dc + dcol] = (__bf16)val;
      } else {
        vT[((size_t)(bb * Hc + h) * Dc + dcol) * Sc + ss] = (__bf16)val;
      }
    }
  }
}

// ---------------------------------------------------------------------------
// Kernel 2: causal flash attention. grid = (B*H, S/128); 8 waves/block, each
// wave owns 16 query rows and streams 32-key chunks (trip count is causal and
// differs per wave -> per-wave LDS slice + s_wait_dscnt, no block barriers).
// ---------------------------------------------------------------------------
__global__ __launch_bounds__(256)
void attn_kernel(const __bf16* __restrict__ q, const __bf16* __restrict__ kbuf,
                 const __bf16* __restrict__ vT, __bf16* __restrict__ merged) {
  __shared__ __align__(32) __bf16 pl[8][16 * 48];  // per-wave P tile, 96B pitch

  const int lane = threadIdx.x & 31;
  const int wv   = threadIdx.x >> 5;
  const int g    = lane >> 4;
  const int ln   = lane & 15;
  const int bh   = blockIdx.x;                  // 0..31
  const int bb   = bh >> 4, hh = bh & 15;
  const int row0 = (blockIdx.y * 8 + wv) * 16;  // query tile start

  const __bf16* qb = q    + (size_t)bh * Sc * Dc;
  const __bf16* kb = kbuf + (size_t)bh * Sc * Dc;
  const __bf16* vb = vT   + (size_t)bh * Dc * Sc;
  __bf16* myp = &pl[wv][0];

  // Resident Q A-fragments (16x32 each) covering D = 0..31 and 32..63.
  v16bf qa[2];
  {
    const __bf16* p = qb + (size_t)(row0 + ln) * Dc;
#pragma unroll
    for (int c = 0; c < 2; ++c) {
      v8bf lo = *(const v8bf*)(p + c * 32 + 16 * g);
      v8bf hi = *(const v8bf*)(p + c * 32 + 16 * g + 8);
#pragma unroll
      for (int i = 0; i < 8; ++i) { qa[c][i] = lo[i]; qa[c][8 + i] = hi[i]; }
    }
  }

  const v8f zero = {0.f, 0.f, 0.f, 0.f, 0.f, 0.f, 0.f, 0.f};
  v8f acc[4] = {zero, zero, zero, zero};
  float m_row[8], l_row[8];
#pragma unroll
  for (int r = 0; r < 8; ++r) { m_row[r] = -3.0e38f; l_row[r] = 0.f; }

  for (int kstart = 0; kstart < row0 + 16; kstart += 32) {
    if (kstart + 32 < row0 + 16)
      __builtin_prefetch(kb + (size_t)(kstart + 32) * Dc, 0, 3);

    // ---- scores for this 32-key chunk: two 16x16 C fragments
    v8f s[2];
#pragma unroll
    for (int t = 0; t < 2; ++t) {
      s[t] = zero;
      const int key = kstart + t * 16 + ln;     // B-frag: lane N = key column
      const __bf16* kr = kb + (size_t)key * Dc + 16 * g;
      v16bf b0 = *(const v16bf*)kr;             // K^T rows d = 16g..16g+15
      v16bf b1 = *(const v16bf*)(kr + 32);      // d = 32+16g..
      s[t] = wmma_bf16(qa[0], b0, s[t]);
      s[t] = wmma_bf16(qa[1], b1, s[t]);
#pragma unroll
      for (int r = 0; r < 8; ++r) {             // causal mask in C layout
        const int qrow = row0 + r + 8 * g;
        if (key > qrow) s[t][r] = -1.0e30f;
      }
    }

    // ---- online softmax (row reductions across the 16-lane half-wave)
#pragma unroll
    for (int r = 0; r < 8; ++r) {
      float mx = fmaxf(s[0][r], s[1][r]);
      mx = fmaxf(mx, __shfl_xor(mx, 1));
      mx = fmaxf(mx, __shfl_xor(mx, 2));
      mx = fmaxf(mx, __shfl_xor(mx, 4));
      mx = fmaxf(mx, __shfl_xor(mx, 8));
      const float mnew = fmaxf(m_row[r], mx);
      const float corr = __expf(m_row[r] - mnew);
      m_row[r] = mnew;
      const float p0 = __expf(s[0][r] - mnew);
      const float p1 = __expf(s[1][r] - mnew);
      s[0][r] = p0; s[1][r] = p1;
      float rs = p0 + p1;
      rs += __shfl_xor(rs, 1);
      rs += __shfl_xor(rs, 2);
      rs += __shfl_xor(rs, 4);
      rs += __shfl_xor(rs, 8);
      l_row[r] = l_row[r] * corr + rs;
#pragma unroll
      for (int f = 0; f < 4; ++f) acc[f][r] *= corr;
    }

    // ---- transpose P: C layout -> A layout via per-wave LDS tile
#pragma unroll
    for (int r = 0; r < 8; ++r) {
      const int rr = r + 8 * g;
      myp[rr * 48 + ln]      = (__bf16)s[0][r];
      myp[rr * 48 + 16 + ln] = (__bf16)s[1][r];
    }
    asm volatile("s_wait_dscnt 0" ::: "memory");

    v16bf pa;
    {
      const __bf16* pr = myp + ln * 48;
      v8bf lo = *(const v8bf*)(pr + 16 * g);
      v8bf hi = *(const v8bf*)(pr + 16 * g + 8);
#pragma unroll
      for (int i = 0; i < 8; ++i) { pa[i] = lo[i]; pa[8 + i] = hi[i]; }
    }

    // ---- acc += P (16x32) x V (32x64), V^T rows are contiguous in s
#pragma unroll
    for (int f = 0; f < 4; ++f) {
      const __bf16* vr = vb + (size_t)(f * 16 + ln) * Sc + kstart + 16 * g;
      v16bf bv = *(const v16bf*)vr;
      acc[f] = wmma_bf16(pa, bv, acc[f]);
    }
  }

  // ---- normalize and write merged [B,S,F] bf16
#pragma unroll
  for (int f = 0; f < 4; ++f) {
#pragma unroll
    for (int r = 0; r < 8; ++r) {
      const int   srow = row0 + r + 8 * g;
      const float o    = acc[f][r] / l_row[r];
      merged[(size_t)(bb * Sc + srow) * Fc + hh * Dc + f * 16 + ln] = (__bf16)o;
    }
  }
}

// ---------------------------------------------------------------------------
// Kernel 3: output projection.  Z[4096,1024] bf16 @ w_proj[1024,1024] fp32
// + b_proj -> out fp32.  Wave tile = 32x64 (2 A-fragments share 4 B-fragments
// -> 8 WMMAs per K-step).  W tile staged transposed (bf16) in LDS per block.
// ---------------------------------------------------------------------------
__global__ __launch_bounds__(256)
void proj_gemm_kernel(const __bf16* __restrict__ A, const float* __restrict__ W,
                      const float* __restrict__ bias, float* __restrict__ out) {
  __shared__ __align__(32) __bf16 wt[64][32];   // [n][k] transposed tile, 4KB

  const int lane = threadIdx.x & 31;
  const int wv   = threadIdx.x >> 5;
  const int g    = lane >> 4;
  const int ln   = lane & 15;
  const int m0   = (blockIdx.y * 8 + wv) * 32;  // 32 rows per wave
  const int n0   = blockIdx.x * 64;

  const __bf16* arow0 = A + (size_t)(m0 + ln) * Fc;
  const __bf16* arow1 = A + (size_t)(m0 + 16 + ln) * Fc;

  const v8f zero = {0.f, 0.f, 0.f, 0.f, 0.f, 0.f, 0.f, 0.f};
  v8f acc[2][4] = {{zero, zero, zero, zero}, {zero, zero, zero, zero}};

  for (int k0 = 0; k0 < Fc; k0 += 32) {
    {  // cooperative staging of W[k0..k0+32)[n0..n0+64) -> LDS (bf16, transposed)
      const int t  = threadIdx.x;
      const int kk = t >> 3;              // 0..31
      const int nc = (t & 7) * 8;         // 0..56 step 8
      const float* wp = W + (size_t)(k0 + kk) * Fc + n0 + nc;
      v4f a = *(const v4f*)wp;
      v4f b = *(const v4f*)(wp + 4);
#pragma unroll
      for (int i = 0; i < 4; ++i) {
        wt[nc + i][kk]     = (__bf16)a[i];
        wt[nc + 4 + i][kk] = (__bf16)b[i];
      }
    }
    __syncthreads();

    if (k0 + 32 < Fc) __builtin_prefetch(arow0 + k0 + 32, 0, 3);

    v16bf af0 = a_frag_bf16(arow0, k0, g);
    v16bf af1 = a_frag_bf16(arow1, k0, g);
#pragma unroll
    for (int sub = 0; sub < 4; ++sub) {
      v16bf bf = *(const v16bf*)&wt[sub * 16 + ln][16 * g];
      acc[0][sub] = wmma_bf16(af0, bf, acc[0][sub]);
      acc[1][sub] = wmma_bf16(af1, bf, acc[1][sub]);
    }
    __syncthreads();
  }

#pragma unroll
  for (int half = 0; half < 2; ++half) {
#pragma unroll
    for (int sub = 0; sub < 4; ++sub) {
      const float bv = bias[n0 + sub * 16 + ln];
#pragma unroll
      for (int r = 0; r < 8; ++r) {
        const int m = m0 + half * 16 + r + 8 * g;
        out[(size_t)m * Fc + n0 + sub * 16 + ln] = acc[half][sub][r] + bv;
      }
    }
  }
}

// ---------------------------------------------------------------------------
extern "C" void kernel_launch(void* const* d_in, const int* in_sizes, int n_in,
                              void* d_out, int out_size, void* d_ws, size_t ws_size,
                              hipStream_t stream) {
  (void)in_sizes; (void)n_in; (void)out_size; (void)ws_size;

  const float* states = (const float*)d_in[0];
  // d_in[1] = causal mask: not needed, causality is computed analytically.
  const float* w_attn = (const float*)d_in[2];
  const float* b_attn = (const float*)d_in[3];
  const float* w_proj = (const float*)d_in[4];
  const float* b_proj = (const float*)d_in[5];
  float* out = (float*)d_out;

  char* ws = (char*)d_ws;
  const size_t SEG = (size_t)8 << 20;   // 8 MB per buffer
  __bf16* q      = (__bf16*)(ws);
  __bf16* kbuf   = (__bf16*)(ws + SEG);
  __bf16* vT     = (__bf16*)(ws + 2 * SEG);
  __bf16* merged = (__bf16*)(ws + 3 * SEG);

  // QKV: M=4096 (32 blocks x 8 waves x 16), N=3072 (48 blocks x 64)
  qkv_gemm_kernel<<<dim3(48, 32), 256, 0, stream>>>(states, w_attn, b_attn,
                                                    q, kbuf, vT);
  // Attention: grid = (B*H, S/128)
  attn_kernel<<<dim3(Bc * Hc, Sc / 128), 256, 0, stream>>>(q, kbuf, vT, merged);
  // Projection: N=1024 (16 blocks x 64), M=4096/256 rows -> 16 blocks
  proj_gemm_kernel<<<dim3(16, 16), 256, 0, stream>>>(merged, w_proj, b_proj, out);
}
